// FTLSTM_88424786690265
// MI455X (gfx1250) — matrix operand order repaired and checked by reading, not compiled, not run on hardware
//
#include <hip/hip_runtime.h>
#include <hip/hip_bf16.h>

typedef __attribute__((ext_vector_type(16))) __bf16 v16bf;
typedef __attribute__((ext_vector_type(8)))  float  v8f;

static constexpr int Bn = 256, DXn = 256, DYn = 256, Hn = 512, Tn = 128, Ln = 2;
static constexpr int KG  = DXn + DYn + Hn;  // 1024 : gate GEMM K
static constexpr int NG  = 3 * Hn;          // 1536 : gate GEMM N
static constexpr int KXY = DXn + DYn;       // 512  : packed xy width
static constexpr int LPAD = 40;             // LDS row stride (bf16 elems), 80 B

// ---------------------------------------------------------------- utilities
__global__ __launch_bounds__(256) void ft_zero(unsigned int* p, int n) {
  int i = blockIdx.x * 256 + threadIdx.x;
  if (i < n) p[i] = 0u;
}

__global__ __launch_bounds__(256) void ft_f32_to_bf16(const float* __restrict__ s,
                                                      __bf16* __restrict__ d, int n) {
  int i = blockIdx.x * 256 + threadIdx.x;
  if (i < n) d[i] = (__bf16)s[i];
}

// inputx/inputy are [B, D, T]; produce time-major bf16 XY[t][b][k], k in [0,512)
__global__ __launch_bounds__(256) void ft_pack_xy(const float* __restrict__ x,
                                                  const float* __restrict__ y,
                                                  __bf16* __restrict__ xy) {
  int i = blockIdx.x * 256 + threadIdx.x;   // over T*B*512 = 2^24
  int t = i >> 17;                          // / (B*KXY)
  int r = i & ((1 << 17) - 1);
  int b = r >> 9;
  int k = r & 511;
  float v = (k < DXn) ? x[((size_t)b * DXn + k) * Tn + t]
                      : y[((size_t)b * DYn + (k - DXn)) * Tn + t];
  xy[i] = (__bf16)v;
}

// ---------------------------------------------------- async copy primitives
__device__ __forceinline__ void async_b128_to_lds(const void* gptr, void* lds) {
  unsigned l = (unsigned)(size_t)lds;  // LDS aperture: offset = addr[31:0]
  asm volatile("global_load_async_to_lds_b128 %0, %1, off"
               :: "v"(l), "v"(gptr) : "memory");
}
__device__ __forceinline__ void wait_async0() {
  asm volatile("s_wait_asynccnt 0x0" ::: "memory");
}

// ------------------------------------------------------------- WMMA GEMM
// out[m,n] = act( sum_k A[m,k] * W[n,k] + bias[n] ),  M fixed = 256.
// A split along K into region1 (A1, K1) + region2 (A2, K2), row-major bf16.
// W is [N, K1+K2] row-major bf16. Two independent problem instances selected
// by blockIdx.z (T-path / F-path share a shape). LDS double-buffered via
// GLOBAL_LOAD_ASYNC_TO_LDS_B128 (ASYNCcnt): tile kc+1 streams into buf^1
// while the 8 WMMAs of tile kc execute; W tile shared by all 8 waves.
struct GemmArgs {
  const __bf16* A1[2];
  const __bf16* A2[2];
  const __bf16* W[2];
  const float*  bias[2];
  float*        out[2];
  int lda1, K1, lda2, K2, N, act;  // act: 0=sigmoid, 1=tanh
};

__global__ __launch_bounds__(256) void ft_gemm_wmma(GemmArgs ga) {
  const int z = blockIdx.z;
  const __bf16* __restrict__ A1 = ga.A1[z];
  const __bf16* __restrict__ A2 = ga.A2[z];
  const __bf16* __restrict__ W  = ga.W[z];
  const float*  __restrict__ bias = ga.bias[z];
  float*        __restrict__ out  = ga.out[z];
  const int K1 = ga.K1, K2 = ga.K2, lda1 = ga.lda1, lda2 = ga.lda2;
  const int N = ga.N, act = ga.act;
  const int K = K1 + K2, nk = K >> 5;

  __shared__ __bf16 shA[2][128][LPAD];  // 20 KB
  __shared__ __bf16 shW[2][128][LPAD];  // 20 KB

  const int tid  = threadIdx.x;
  const int lane = tid & 31;
  const int wave = tid >> 5;
  const int m0   = blockIdx.y * 128;
  const int n0   = blockIdx.x * 128;

  // loader mapping: 512 (row, 8-elem seg) pieces per tile; 2 per thread
  const int r0 = tid >> 2, s0 = (tid & 3) * 8;
  const int r1 = r0 + 64,  s1 = s0;

  // stream one 128x32 A tile + 128x32 W tile straight into LDS (no VGPRs)
  auto issue = [&](int kc, int buf) {
    const int kb = kc << 5;
    const __bf16* ab; int ld, kloc;
    if (kb < K1) { ab = A1; ld = lda1; kloc = kb; }
    else         { ab = A2; ld = lda2; kloc = kb - K1; }
    async_b128_to_lds(ab + (size_t)(m0 + r0) * ld + kloc + s0, &shA[buf][r0][s0]);
    async_b128_to_lds(ab + (size_t)(m0 + r1) * ld + kloc + s1, &shA[buf][r1][s1]);
    async_b128_to_lds(W + (size_t)(n0 + r0) * K + kb + s0,     &shW[buf][r0][s0]);
    async_b128_to_lds(W + (size_t)(n0 + r1) * K + kb + s1,     &shW[buf][r1][s1]);
  };

  // fragment geometry (ISA 16-bit layouts, wave32)
  const int arow = wave * 16 + (lane & 15);
  const int aoff = (lane & 16) ? 8 : 0;
  const int boff = (lane & 16) ? 8 : 0;
  const int nl   = lane & 15;

  v8f acc[8];
#pragma unroll
  for (int i = 0; i < 8; ++i) acc[i] = (v8f){0.f, 0.f, 0.f, 0.f, 0.f, 0.f, 0.f, 0.f};

  issue(0, 0);
  wait_async0();
  __syncthreads();  // tile 0 resident for all waves

  for (int kc = 0; kc < nk; ++kc) {
    const int cur = kc & 1;
    if (kc + 1 < nk) issue(kc + 1, cur ^ 1);  // overlap next tile with WMMAs

    union { v16bf v; uint4 q[2]; } af;
    af.q[0] = *(const uint4*)&shA[cur][arow][aoff];
    af.q[1] = *(const uint4*)&shA[cur][arow][aoff + 16];
#pragma unroll
    for (int nt = 0; nt < 8; ++nt) {
      union { v16bf v; uint4 q[2]; } bf_;
      bf_.q[0] = *(const uint4*)&shW[cur][nt * 16 + nl][boff];
      bf_.q[1] = *(const uint4*)&shW[cur][nt * 16 + nl][boff + 16];
      acc[nt] = __builtin_amdgcn_wmma_f32_16x16x32_bf16(
          false, af.v, false, bf_.v, (short)0, acc[nt], false, false);
    }

    if (kc + 1 < nk) {
      __syncthreads();   // all waves done reading buf `cur`
      wait_async0();     // my async transfers for tile kc+1 have landed
      __syncthreads();   // everyone's tile kc+1 transfers have landed
    }
  }

  // epilogue: bias + activation, f32 store. D layout: VGPR r -> M = +r (+8 hi lanes)
  const int mbase = m0 + wave * 16 + ((lane & 16) ? 8 : 0);
#pragma unroll
  for (int nt = 0; nt < 8; ++nt) {
    const int n = n0 + nt * 16 + nl;
    const float bv = bias[n];
#pragma unroll
    for (int r = 0; r < 8; ++r) {
      float v = acc[nt][r] + bv;
      v = (act == 0) ? (1.0f / (1.0f + __expf(-v))) : tanhf(v);
      out[(size_t)(mbase + r) * N + n] = v;
    }
  }
}

// ------------------------------------------------------- pointwise LSTM cell
// gates [B, 3H] (f,i,o already sigmoided), cand [B,H] already tanh'd.
__global__ __launch_bounds__(256) void ft_lstm_update(
    const float* __restrict__ gates, const float* __restrict__ cand,
    float* __restrict__ C, float* __restrict__ hf, __bf16* __restrict__ hb) {
  int i = blockIdx.x * 256 + threadIdx.x;   // B*H
  int b = i >> 9, c = i & 511;
  const float* g = gates + (size_t)b * NG;
  float f  = g[c];
  float ig = g[Hn + c];
  float o  = g[2 * Hn + c];
  float cn = f * C[i] + ig * cand[i];
  float h  = o * tanhf(cn);
  C[i]  = cn;
  hf[i] = h;
  hb[i] = (__bf16)h;
}

// ----------------------------------------------- train-mode BN over batch dim
// one block per (channel, path); 256 threads = batch. Writes out[b, c, t].
__global__ __launch_bounds__(256) void ft_bn_out(
    const float* __restrict__ hT, const float* __restrict__ hF,
    const float* __restrict__ gT, const float* __restrict__ bT,
    const float* __restrict__ gF, const float* __restrict__ bF,
    float* __restrict__ outbuf, int t) {
  const int c = blockIdx.x;
  const int path = blockIdx.y;
  const int b = threadIdx.x;
  const float* h = path ? hF : hT;
  float v = h[(size_t)b * Hn + c];
  __shared__ float s1[256];
  __shared__ float s2[256];
  s1[b] = v;
  s2[b] = v * v;
  __syncthreads();
  for (int st = 128; st > 0; st >>= 1) {
    if (b < st) { s1[b] += s1[b + st]; s2[b] += s2[b + st]; }
    __syncthreads();
  }
  float mu  = s1[0] * (1.0f / Bn);
  float var = s2[0] * (1.0f / Bn) - mu * mu;   // biased variance
  float ga  = path ? gF[c] : gT[c];
  float be  = path ? bF[c] : bT[c];
  float o   = (v - mu) * rsqrtf(var + 1e-5f) * ga + be;
  outbuf[(size_t)path * Bn * Hn * Tn + (size_t)b * Hn * Tn + (size_t)c * Tn + t] = o;
}

// ------------------------------------------------------------------- driver
extern "C" void kernel_launch(void* const* d_in, const int* in_sizes, int n_in,
                              void* d_out, int out_size, void* d_ws, size_t ws_size,
                              hipStream_t stream) {
  (void)in_sizes; (void)n_in; (void)out_size; (void)ws_size;
  const float* inputx = (const float*)d_in[0];
  const float* inputy = (const float*)d_in[1];
  const float* WTg    = (const float*)d_in[2];
  const float* bTg    = (const float*)d_in[3];
  const float* WFg    = (const float*)d_in[4];
  const float* bFg    = (const float*)d_in[5];
  const float* WTc    = (const float*)d_in[6];
  const float* bTc    = (const float*)d_in[7];
  const float* WFc    = (const float*)d_in[8];
  const float* bFc    = (const float*)d_in[9];
  const float* gammaT = (const float*)d_in[10];
  const float* betaT  = (const float*)d_in[11];
  const float* gammaF = (const float*)d_in[12];
  const float* betaF  = (const float*)d_in[13];
  float* out = (float*)d_out;
  char*  ws  = (char*)d_ws;

  const int l = Ln - 1;  // only the last layer reaches the output

  size_t off = 0;
  auto take = [&](size_t bytes) {
    size_t r = off;
    off += (bytes + 255) & ~(size_t)255;
    return r;
  };
  __bf16* XY     = (__bf16*)(ws + take((size_t)Tn * Bn * KXY * 2));
  __bf16* WgT_bf = (__bf16*)(ws + take((size_t)NG * KG * 2));
  __bf16* WgF_bf = (__bf16*)(ws + take((size_t)NG * KG * 2));
  __bf16* WcT_bf = (__bf16*)(ws + take((size_t)Hn * (DXn + Hn) * 2));
  __bf16* WcF_bf = (__bf16*)(ws + take((size_t)Hn * (DYn + Hn) * 2));
  char* zbase = ws + off;  // [hbT, hbF, CT, CF] zeroed each call
  __bf16* hbT = (__bf16*)(ws + take((size_t)Bn * Hn * 2));
  __bf16* hbF = (__bf16*)(ws + take((size_t)Bn * Hn * 2));
  float*  CT  = (float*)(ws + take((size_t)Bn * Hn * 4));
  float*  CF  = (float*)(ws + take((size_t)Bn * Hn * 4));
  size_t zwords = ((ws + off) - zbase) / 4;
  float* hfT    = (float*)(ws + take((size_t)Bn * Hn * 4));
  float* hfF    = (float*)(ws + take((size_t)Bn * Hn * 4));
  float* gatesT = (float*)(ws + take((size_t)Bn * NG * 4));
  float* gatesF = (float*)(ws + take((size_t)Bn * NG * 4));
  float* candT  = (float*)(ws + take((size_t)Bn * Hn * 4));
  float* candF  = (float*)(ws + take((size_t)Bn * Hn * 4));

  dim3 blk(256);

  // one-time prep (per call; deterministic)
  ft_zero<<<(int)((zwords + 255) / 256), blk, 0, stream>>>((unsigned int*)zbase, (int)zwords);
  const int nWg = NG * KG;
  ft_f32_to_bf16<<<(nWg + 255) / 256, blk, 0, stream>>>(WTg + (size_t)l * nWg, WgT_bf, nWg);
  ft_f32_to_bf16<<<(nWg + 255) / 256, blk, 0, stream>>>(WFg + (size_t)l * nWg, WgF_bf, nWg);
  const int nWc = Hn * (DXn + Hn);
  ft_f32_to_bf16<<<(nWc + 255) / 256, blk, 0, stream>>>(WTc + (size_t)l * nWc, WcT_bf, nWc);
  ft_f32_to_bf16<<<(nWc + 255) / 256, blk, 0, stream>>>(WFc + (size_t)l * nWc, WcF_bf, nWc);
  const int nXY = Tn * Bn * KXY;
  ft_pack_xy<<<nXY / 256, blk, 0, stream>>>(inputx, inputy, XY);

  const float* bgT = bTg + (size_t)l * NG;
  const float* bgF = bFg + (size_t)l * NG;
  const float* bcT = bTc + (size_t)l * Hn;
  const float* bcF = bFc + (size_t)l * Hn;
  const float* gaT = gammaT + (size_t)l * Hn;
  const float* beT = betaT + (size_t)l * Hn;
  const float* gaF = gammaF + (size_t)l * Hn;
  const float* beF = betaF + (size_t)l * Hn;

  const dim3 gGate(NG / 128, Bn / 128, 2);  // (12, 2, 2) — z: T-path / F-path
  const dim3 gCand(Hn / 128, Bn / 128, 2);  // (4, 2, 2)

  for (int t = 0; t < Tn; ++t) {
    const __bf16* xyt = XY + (size_t)t * Bn * KXY;

    // gates: sigmoid([x;y;h] @ Wg^T + b), K = 512 (xy) + 512 (h), both paths
    GemmArgs ag;
    ag.A1[0] = xyt;    ag.A1[1] = xyt;
    ag.A2[0] = hbT;    ag.A2[1] = hbF;
    ag.W[0]  = WgT_bf; ag.W[1]  = WgF_bf;
    ag.bias[0] = bgT;  ag.bias[1] = bgF;
    ag.out[0] = gatesT; ag.out[1] = gatesF;
    ag.lda1 = KXY; ag.K1 = KXY; ag.lda2 = Hn; ag.K2 = Hn; ag.N = NG; ag.act = 0;
    ft_gemm_wmma<<<gGate, blk, 0, stream>>>(ag);

    // candidates: tanh([x;h] / [y;h] @ Wc^T + b), K = 256 + 512, both paths
    GemmArgs ac;
    ac.A1[0] = xyt;    ac.A1[1] = xyt + DXn;
    ac.A2[0] = hbT;    ac.A2[1] = hbF;
    ac.W[0]  = WcT_bf; ac.W[1]  = WcF_bf;
    ac.bias[0] = bcT;  ac.bias[1] = bcF;
    ac.out[0] = candT; ac.out[1] = candF;
    ac.lda1 = KXY; ac.K1 = DXn; ac.lda2 = Hn; ac.K2 = Hn; ac.N = Hn; ac.act = 1;
    ft_gemm_wmma<<<gCand, blk, 0, stream>>>(ac);

    // cell update
    ft_lstm_update<<<Bn * Hn / 256, blk, 0, stream>>>(gatesT, candT, CT, hfT, hbT);
    ft_lstm_update<<<Bn * Hn / 256, blk, 0, stream>>>(gatesF, candF, CF, hfF, hbF);
    // batchnorm + scatter into [B, H, T] (both paths)
    ft_bn_out<<<dim3(Hn, 2), blk, 0, stream>>>(hfT, hfF, gaT, beT, gaF, beF, out, t);
  }
}
